// AdvisorCrossAttentionAdapter_38766374814335
// MI455X (gfx1250) — matrix operand order repaired and checked by reading, not compiled, not run on hardware
//
#include <hip/hip_runtime.h>

// ---------------- CDNA5 WMMA types ----------------
typedef __attribute__((ext_vector_type(16))) __bf16 v16bf;
typedef __attribute__((ext_vector_type(8)))  __bf16 v8bf;
typedef __attribute__((ext_vector_type(8)))  float  v8f;
typedef __attribute__((ext_vector_type(4)))  float  v4f;

// Problem sizes (fixed by the reference)
static constexpr long B_  = 4;
static constexpr long S_  = 2048;   // query length
static constexpr long LA_ = 3072;   // advisor length
static constexpr long T_  = 1024;   // LA/3
static constexpr long H_  = 2048;   // hidden

// ---------------- f32 -> bf16 convert (8 elems / thread) ----------------
__global__ void f32_to_bf16_kernel(const float* __restrict__ in,
                                   __bf16* __restrict__ out, long n8) {
    long i = (long)blockIdx.x * blockDim.x + threadIdx.x;
    if (i >= n8) return;
    v4f a = *(const v4f*)(in + i * 8);
    v4f b = *(const v4f*)(in + i * 8 + 4);
    v8bf o;
#pragma unroll
    for (int j = 0; j < 4; ++j) { o[j] = (__bf16)a[j]; o[4 + j] = (__bf16)b[j]; }
    *(v8bf*)(out + i * 8) = o;
}

// ---------------- Register-blocked bf16 WMMA GEMM ----------------
// C[m][n] = sum_k A[rowAdd + m*rowMul][k] * B[n][k]
// A row-major (lda); B row-major over n (ldb): column n of the WMMA B-matrix
// is row n of the array -> contiguous per-lane K runs (no transpose needed).
// Each wave computes a 32x64 tile: 2 M-subtiles x 4 N-subtiles, sharing
// A fragments across 4 WMMAs and B fragments across 2 WMMAs.
// Block = 4 waves -> block tile 32 x 256.
__global__ __launch_bounds__(128)
void gemm_bf16_wmma(const __bf16* __restrict__ A, long aBatch, int lda,
                    int rowMul, int rowAdd,
                    const __bf16* __restrict__ Bm, long bBatch, int ldb,
                    void* __restrict__ C, long cBatch, int ldc, int cIsF32,
                    int K) {
    const int lane = threadIdx.x & 31;
    const int wave = threadIdx.x >> 5;
    const int n0   = blockIdx.x * 256 + wave * 64;
    const int m0   = blockIdx.y * 32;
    const int bz   = blockIdx.z;

    const __bf16* Ab = A  + (long)bz * aBatch;
    const __bf16* Bb = Bm + (long)bz * bBatch;

    // A fragment: lane<16 -> row M=lane,    K halves {0..7, 16..23}
    //             lane>=16-> row M=lane-16, K halves {8..15, 24..31}
    const int koff = (lane < 16) ? 0 : 8;
    const __bf16* arow0 = Ab + (long)((m0 + (lane & 15)) * rowMul + rowAdd) * lda + koff;
    const __bf16* arow1 = arow0 + (long)16 * rowMul * lda;

    // B fragment: lane<16 -> col n+lane, K {0..7, 16..23} ; lane>=16 -> col n+lane-16, K {8..15, 24..31}
    const __bf16* bcol = Bb + (long)(n0 + (lane & 15)) * ldb + ((lane < 16) ? 0 : 8);
    const long bstep = (long)16 * ldb;   // next 16-column group

    v8f acc[2][4] = {};
    union AFrag { v16bf v; v8bf h[2]; };

#pragma unroll 2
    for (int k0 = 0; k0 < K; k0 += 32) {
        AFrag a0, a1;
        a0.h[0] = *(const v8bf*)(arow0 + k0);
        a0.h[1] = *(const v8bf*)(arow0 + k0 + 16);
        a1.h[0] = *(const v8bf*)(arow1 + k0);
        a1.h[1] = *(const v8bf*)(arow1 + k0 + 16);
        v16bf bf[4];
#pragma unroll
        for (int j = 0; j < 4; ++j) {
            AFrag t;
            t.h[0] = *(const v8bf*)(bcol + j * bstep + k0);
            t.h[1] = *(const v8bf*)(bcol + j * bstep + k0 + 16);
            bf[j] = t.v;
        }
#pragma unroll
        for (int j = 0; j < 4; ++j) {
            acc[0][j] = __builtin_amdgcn_wmma_f32_16x16x32_bf16(
                false, a0.v, false, bf[j], (short)0, acc[0][j], false, false);
            acc[1][j] = __builtin_amdgcn_wmma_f32_16x16x32_bf16(
                false, a1.v, false, bf[j], (short)0, acc[1][j], false, false);
        }
    }

    // C layout: lane holds column (lane&15) of its N-subtile;
    // VGPR r -> row r (lanes 0-15) or row 8+r (lanes 16-31)
    const int rbase = (lane < 16) ? 0 : 8;
    if (cIsF32) {
        float* Cf = (float*)C + (long)bz * cBatch;
#pragma unroll
        for (int i = 0; i < 2; ++i)
#pragma unroll
            for (int j = 0; j < 4; ++j) {
                const int ncol = n0 + 16 * j + (lane & 15);
#pragma unroll
                for (int r = 0; r < 8; ++r)
                    Cf[(long)(m0 + 16 * i + rbase + r) * ldc + ncol] = acc[i][j][r];
            }
    } else {
        __bf16* Cb = (__bf16*)C + (long)bz * cBatch;
#pragma unroll
        for (int i = 0; i < 2; ++i)
#pragma unroll
            for (int j = 0; j < 4; ++j) {
                const int ncol = n0 + 16 * j + (lane & 15);
#pragma unroll
                for (int r = 0; r < 8; ++r)
                    Cb[(long)(m0 + 16 * i + rbase + r) * ldc + ncol] = (__bf16)acc[i][j][r];
            }
    }
}

// ---------------- Gate + transpose-store of V ----------------
// v_rel = slot0 proj, v1 = slot1 proj, v2 = slot2 proj; id = ids[b][3t]
// Writes vT[b][h][T] so the ctx GEMM gets contiguous B-fragment columns.
__global__ void gate_kernel(const __bf16* __restrict__ v0,
                            const __bf16* __restrict__ v1,
                            const __bf16* __restrict__ v2,
                            const int* __restrict__ ids,
                            __bf16* __restrict__ vT) {
    const int hcol = blockIdx.x * blockDim.x + threadIdx.x;
    const int t    = blockIdx.y;
    const int b    = blockIdx.z;
    const long idx = ((long)(b * T_ + t)) * H_ + hcol;
    const float vr = (float)v0[idx];
    const float x  = (float)v1[idx];
    const float y  = (float)v2[idx];
    const int  id  = ids[(long)b * LA_ + 3L * t];
    float r;
    if      (id == 0) r = fminf(x, y);          // and
    else if (id == 1) r = fmaxf(x, y);          // or
    else if (id == 2) r = -x;                   // not
    else if (id == 3) r = fmaxf(-x, y);         // implies
    else if (id == 4) r = fabsf(x - y);         // xor
    else              r = vr;                   // learned (id >= 5)
    vT[((long)(b * H_ + hcol)) * T_ + t] = (__bf16)r;
}

// ---------------- Row softmax (scale 1/sqrt(h)), f32 -> bf16 P ----------------
__global__ void softmax_rows_kernel(const float* __restrict__ sc,
                                    __bf16* __restrict__ p, float scale) {
    __shared__ float red[256];
    const long row = blockIdx.x;            // b*S + s
    const float* s = sc + row * T_;
    __bf16* pr = p + row * T_;
    const int tid = threadIdx.x;

    float m = -3.0e38f;
    for (int i = tid; i < T_; i += 256) m = fmaxf(m, s[i] * scale);
    red[tid] = m; __syncthreads();
    for (int off = 128; off > 0; off >>= 1) {
        if (tid < off) red[tid] = fmaxf(red[tid], red[tid + off]);
        __syncthreads();
    }
    m = red[0]; __syncthreads();

    float sum = 0.f;
    for (int i = tid; i < T_; i += 256) sum += __expf(s[i] * scale - m);
    red[tid] = sum; __syncthreads();
    for (int off = 128; off > 0; off >>= 1) {
        if (tid < off) red[tid] += red[tid + off];
        __syncthreads();
    }
    const float inv = 1.0f / red[0];
    for (int i = tid; i < T_; i += 256)
        pr[i] = (__bf16)(__expf(s[i] * scale - m) * inv);
}

// ---------------- Host side ----------------
extern "C" void kernel_launch(void* const* d_in, const int* in_sizes, int n_in,
                              void* d_out, int out_size, void* d_ws, size_t ws_size,
                              hipStream_t stream) {
    const float* hidden  = (const float*)d_in[0];   // [B,S,H] f32
    const float* advisor = (const float*)d_in[1];   // [B,LA,H] f32
    const int*   ids     = (const int*)d_in[2];     // [B,LA] int32 (jax default x64=off)
    const float* Wq      = (const float*)d_in[3];   // [H,H]
    const float* Wk      = (const float*)d_in[4];
    const float* Wv      = (const float*)d_in[5];
    const float* Wo      = (const float*)d_in[6];

    // --- workspace carve-out (bytes, 256B aligned) ---
    char* ws = (char*)d_ws;
    size_t off = 0;
    auto carve = [&](size_t bytes) {
        char* p = ws + off;
        off += (bytes + 255) & ~(size_t)255;
        return p;
    };
    __bf16* hb  = (__bf16*)carve(2 * B_ * S_  * H_);   // hidden bf16
    __bf16* ab  = (__bf16*)carve(2 * B_ * LA_ * H_);   // advisor bf16
    __bf16* wqb = (__bf16*)carve(2 * H_ * H_);
    __bf16* wkb = (__bf16*)carve(2 * H_ * H_);
    __bf16* wvb = (__bf16*)carve(2 * H_ * H_);
    __bf16* wob = (__bf16*)carve(2 * H_ * H_);
    __bf16* qb  = (__bf16*)carve(2 * B_ * S_ * H_);    // q
    __bf16* kb  = (__bf16*)carve(2 * B_ * T_ * H_);    // k_final
    __bf16* v0b = (__bf16*)carve(2 * B_ * T_ * H_);    // v slot0 proj
    __bf16* v1b = (__bf16*)carve(2 * B_ * T_ * H_);
    __bf16* v2b = (__bf16*)carve(2 * B_ * T_ * H_);
    __bf16* vTb = (__bf16*)carve(2 * B_ * H_ * T_);    // gated V, transposed
    float*  sc  = (float*) carve(4 * B_ * S_ * T_);    // raw scores f32
    __bf16* pb  = (__bf16*)carve(2 * B_ * S_ * T_);    // softmax probs bf16
    __bf16* ctx = (__bf16*)carve(2 * B_ * S_ * H_);    // attention context

    // --- 1) convert everything to bf16 (vectorized x8) ---
    auto cvt = [&](const float* src, __bf16* dst, long n) {
        long n8 = n / 8;
        f32_to_bf16_kernel<<<dim3((unsigned)((n8 + 255) / 256)), dim3(256), 0, stream>>>(src, dst, n8);
    };
    cvt(hidden,  hb,  B_ * S_  * H_);
    cvt(advisor, ab,  B_ * LA_ * H_);
    cvt(Wq, wqb, H_ * H_);
    cvt(Wk, wkb, H_ * H_);
    cvt(Wv, wvb, H_ * H_);
    cvt(Wo, wob, H_ * H_);

    const dim3 blk(128);  // 4 waves; block tile = 32 rows x 256 cols

    // --- 2) q = hidden @ Wq^T : M=S, N=H, K=H per batch ---
    gemm_bf16_wmma<<<dim3(H_ / 256, S_ / 32, B_), blk, 0, stream>>>(
        hb, S_ * H_, (int)H_, 1, 0, wqb, 0, (int)H_, qb, S_ * H_, (int)H_, 0, (int)H_);

    // --- 3) k = triplets[:,:,0,:] @ Wk^T : M=T (A rows 3t), N=H, K=H ---
    gemm_bf16_wmma<<<dim3(H_ / 256, T_ / 32, B_), blk, 0, stream>>>(
        ab, LA_ * H_, (int)H_, 3, 0, wkb, 0, (int)H_, kb, T_ * H_, (int)H_, 0, (int)H_);

    // --- 4) v_proj slots 0/1/2 : A rows 3t+slot ---
    gemm_bf16_wmma<<<dim3(H_ / 256, T_ / 32, B_), blk, 0, stream>>>(
        ab, LA_ * H_, (int)H_, 3, 0, wvb, 0, (int)H_, v0b, T_ * H_, (int)H_, 0, (int)H_);
    gemm_bf16_wmma<<<dim3(H_ / 256, T_ / 32, B_), blk, 0, stream>>>(
        ab, LA_ * H_, (int)H_, 3, 1, wvb, 0, (int)H_, v1b, T_ * H_, (int)H_, 0, (int)H_);
    gemm_bf16_wmma<<<dim3(H_ / 256, T_ / 32, B_), blk, 0, stream>>>(
        ab, LA_ * H_, (int)H_, 3, 2, wvb, 0, (int)H_, v2b, T_ * H_, (int)H_, 0, (int)H_);

    // --- 5) gate + transposed store vT[b][h][t] ---
    gate_kernel<<<dim3(H_ / 256, T_, B_), dim3(256), 0, stream>>>(v0b, v1b, v2b, ids, vTb);

    // --- 6) scores = q @ k^T : M=S, N=T, K=H, f32 out ---
    gemm_bf16_wmma<<<dim3(T_ / 256, S_ / 32, B_), blk, 0, stream>>>(
        qb, S_ * H_, (int)H_, 1, 0, kb, T_ * H_, (int)H_, sc, S_ * T_, (int)T_, 1, (int)H_);

    // --- 7) softmax over T with scale 1/sqrt(H), bf16 out ---
    softmax_rows_kernel<<<dim3((unsigned)(B_ * S_)), dim3(256), 0, stream>>>(
        sc, pb, 1.0f / sqrtf((float)H_));

    // --- 8) ctx = P @ V : M=S, N=H, K=T ; B = vT[b][h][t] contiguous over t ---
    gemm_bf16_wmma<<<dim3(H_ / 256, S_ / 32, B_), blk, 0, stream>>>(
        pb, S_ * T_, (int)T_, 1, 0, vTb, H_ * T_, (int)T_, ctx, S_ * H_, (int)H_, 0, (int)T_);

    // --- 9) out = ctx @ Wo^T : M=S, N=H, K=H, f32 out to d_out ---
    gemm_bf16_wmma<<<dim3(H_ / 256, S_ / 32, B_), blk, 0, stream>>>(
        ctx, S_ * H_, (int)H_, 1, 0, wob, 0, (int)H_, d_out, S_ * H_, (int)H_, 1, (int)H_);
}
